// MultiheadAttentionXL_43241730736312
// MI455X (gfx1250) — matrix operand order repaired and checked by reading, not compiled, not run on hardware
//
#include <hip/hip_runtime.h>
#include <hip/hip_bf16.h>

typedef unsigned int u32;
typedef __attribute__((ext_vector_type(4)))  u32    u32x4;
typedef __attribute__((ext_vector_type(16))) __bf16 v16bf;
typedef __attribute__((ext_vector_type(8)))  float  v8f;

#define TMEM 1024

#if __has_builtin(__builtin_amdgcn_global_load_async_to_lds_b128)
#define ASYNC_COPY 1
#else
#define ASYNC_COPY 0
#endif

#if __has_builtin(__builtin_amdgcn_tensor_load_to_lds)
#define HAVE_TDM 1
#else
#define HAVE_TDM 0
#endif

// 16-byte global -> LDS copy; async DMA path on CDNA5 (ASYNCcnt), fallback vector copy.
__device__ __forceinline__ void async_cp16(void* l, const void* g) {
#if ASYNC_COPY
  typedef int v4i_t __attribute__((__vector_size__(4 * sizeof(int))));
  typedef __attribute__((address_space(1))) v4i_t* gp_t;
  typedef __attribute__((address_space(3))) v4i_t* lp_t;
  __builtin_amdgcn_global_load_async_to_lds_b128((gp_t)(void*)g, (lp_t)l, 0, 0);
#else
  *(u32x4*)l = *(const u32x4*)g;
#endif
}

__device__ __forceinline__ void async_wait() {
#if ASYNC_COPY
  asm volatile("s_wait_asynccnt 0" ::: "memory");
#endif
}

// ---------------- Tensor Data Mover (TDM) 2D tile load ----------------
typedef unsigned int tdm_u32x4 __attribute__((__vector_size__(16)));
typedef int          tdm_i32x8 __attribute__((__vector_size__(32)));
typedef int          tdm_i32x4 __attribute__((__vector_size__(16)));

__device__ __forceinline__ u32 lds_addr_of(const void* p) {
  return (u32)(unsigned long long)(__attribute__((address_space(3))) const void*)p;
}

// All dims/strides in 8-byte units (data_size = 3). 2D tile of a 2D tensor.
// D# layout per CDNA5 ISA 8.3/8.4: group0 = {flags(count=1), lds_addr,
// global_addr[31:0], global_addr[56:32] | type=2<<30}; group1 packs data_size,
// tensor_dim0/1, tile_dim0/1, tensor_dim0_stride. OOB reads return zero.
__device__ __forceinline__ void tdm_load_2d(const void* g, u32 lds,
    u32 tile_d0, u32 tile_d1, u32 ten_d0, u32 ten_d1, unsigned long long stride0) {
#if HAVE_TDM
  unsigned long long ga = (unsigned long long)g;
  tdm_u32x4 g0 = { 1u, lds, (u32)ga,
                   (u32)(((ga >> 32) & 0x01FFFFFFull) | 0x80000000u) };
  tdm_i32x8 g1 = {
    (int)(3u << 16),                                        // data_size=8B, mask=0
    (int)((ten_d0 & 0xFFFFu) << 16),                        // abar=0 | tensor_dim0 lo16
    (int)((ten_d0 >> 16) | ((ten_d1 & 0xFFFFu) << 16)),     // dim0 hi16 | dim1 lo16
    (int)((ten_d1 >> 16) | (tile_d0 << 16)),                // dim1 hi16 | tile_dim0
    (int)tile_d1,                                           // tile_dim1 | tile_dim2=0
    (int)(u32)(stride0 & 0xFFFFFFFFull),                    // dim0_stride lo32
    (int)(u32)((stride0 >> 32) & 0xFFFFull),                // dim0_stride hi16 | d1stride=0
    0
  };
  tdm_i32x4 z4 = { 0, 0, 0, 0 };
  tdm_i32x8 z8 = { 0, 0, 0, 0, 0, 0, 0, 0 };
  __builtin_amdgcn_tensor_load_to_lds(g0, g1, z4, z4, z8, 0);
#else
  (void)g; (void)lds; (void)tile_d0; (void)tile_d1;
  (void)ten_d0; (void)ten_d1; (void)stride0;
#endif
}

__device__ __forceinline__ void tdm_wait() {
#if HAVE_TDM
  __builtin_amdgcn_s_wait_tensorcnt(0);
#endif
}

__device__ __forceinline__ v8f wmma_bf16(v16bf a, v16bf b, v8f c) {
  return __builtin_amdgcn_wmma_f32_16x16x32_bf16(false, a, false, b, (short)0, c, false, false);
}

// A-fragment (16x32 bf16, MxK). Lane l: M = l&15, half = l>>4.
__device__ __forceinline__ v16bf ld_fragA(const __bf16* rowptr, int half, int koff) {
  union { v16bf v; u32 u[8]; } f;
  const u32* p = reinterpret_cast<const u32*>(rowptr);
  int b0 = (koff >> 1) + half * 8;
#pragma unroll
  for (int t = 0; t < 8; ++t) f.u[t] = p[b0 + t];
  return f.v;
}

// B-fragment (32x16 bf16, KxN), LDS stored [n][k] contiguous in k.
__device__ __forceinline__ v16bf ld_fragB(const __bf16* rowptr, int half, int koff) {
  union { v16bf v; u32 u[8]; } f;
  const u32* p = reinterpret_cast<const u32*>(rowptr) + (koff >> 1) + half * 8;
#pragma unroll
  for (int t = 0; t < 8; ++t) f.u[t] = p[t];
  return f.v;
}

// ---------------- elementwise convert ----------------
__global__ __launch_bounds__(256) void cvt_kernel(const float* __restrict__ src,
                                                  __bf16* __restrict__ dst, int n) {
  int i = blockIdx.x * 256 + threadIdx.x;
  if (i < n) dst[i] = (__bf16)src[i];
}

// transpose convert: dst[n][k] = src[k][n]; rows = K-dim, cols = N-dim
__global__ __launch_bounds__(256) void cvt_t_kernel(const float* __restrict__ src,
                                                    __bf16* __restrict__ dst,
                                                    int rows, int cols) {
  int i = blockIdx.x * 256 + threadIdx.x;        // i = n*rows + k (coalesced writes)
  if (i < rows * cols) {
    int k = i % rows;
    int n = i / rows;
    dst[i] = (__bf16)src[(size_t)k * cols + n];
  }
}

// cat = concat(mems, inpts) over time axis, bf16, rows = kk*B+b
__global__ __launch_bounds__(256) void cat_kernel(const float* __restrict__ mems,
                                                  const float* __restrict__ inpts,
                                                  __bf16* __restrict__ cat) {
  int i = blockIdx.x * 256 + threadIdx.x;        // over 4096*1024
  int d = i & 1023;
  int rb = i >> 10;
  int b = rb & 1;
  int kk = rb >> 1;
  float v = (kk < TMEM) ? mems[((size_t)kk * 2 + b) * 1024 + d]
                        : inpts[((size_t)(kk - TMEM) * 2 + b) * 1024 + d];
  cat[i] = (__bf16)v;
}

// ---------------- generic bf16 WMMA GEMM (Kc = 1024, B pre-transposed) ----------------
__device__ __forceinline__ void gemm_epi(int mode, int row, int col, float v,
    const float* rwb, const float* rrb,
    __bf16* qa, __bf16* qb, __bf16* ka, __bf16* va, __bf16* ra, float* aout) {
  if (mode == 0) {                               // w_heads: scatter q/k/v
    int kk = row >> 1, b = row & 1;
    int sec = col >> 10, cc = col & 1023, n = cc >> 6, d = cc & 63;
    if (sec == 0) {
      if (kk >= TMEM) {
        int i = kk - TMEM;
        size_t o = ((size_t)(b * 16 + n) * 1024 + i) * 64 + d;
        qa[o] = (__bf16)(v + rwb[n * 64 + d]);
        qb[o] = (__bf16)(v + rrb[n * 64 + d]);
      }
    } else if (sec == 1) {
      ka[((size_t)(b * 16 + n) * 2048 + kk) * 64 + d] = (__bf16)v;
    } else {                                     // V transposed: [b][n][d][j]
      va[((size_t)(b * 16 + n) * 64 + d) * 2048 + kk] = (__bf16)v;
    }
  } else if (mode == 1) {                        // r_head_k -> [n][m][d]
    int n = col >> 6, d = col & 63;
    ra[((size_t)n * 2048 + row) * 64 + d] = (__bf16)v;
  } else {                                       // attn_out fp32
    aout[(size_t)row * 1024 + col] = v;
  }
}

__global__ __launch_bounds__(128) void gemm_bf16_kernel(
    const __bf16* __restrict__ A, const __bf16* __restrict__ Bt, int mode,
    const float* __restrict__ rwb, const float* __restrict__ rrb,
    __bf16* qa, __bf16* qb, __bf16* ka, __bf16* va, __bf16* ra, float* aout) {
  __shared__ __bf16 sA[64 * 32];                 // [m][k]
  __shared__ __bf16 sB[64 * 32];                 // [n][k]
  const int Kc = 1024;
  int tid = threadIdx.x, wave = tid >> 5, lane = tid & 31, lr = lane & 15, half = lane >> 4;
  int c0 = blockIdx.x * 64, r0 = blockIdx.y * 64;
  int wr = (wave >> 1) * 32, wc = (wave & 1) * 32;
  v8f a00 = {0,0,0,0,0,0,0,0}, a01 = a00, a10 = a00, a11 = a00;
  for (int k0 = 0; k0 < Kc; k0 += 32) {
    __syncthreads();
#pragma unroll
    for (int t = 0; t < 2; ++t) {                // A: 256 x 16B chunks
      int c = tid + t * 128;
      int row = c >> 2, cc = c & 3;
      async_cp16(sA + c * 8, A + (size_t)(r0 + row) * Kc + k0 + cc * 8);
    }
#pragma unroll
    for (int t = 0; t < 2; ++t) {                // B: 256 x 16B chunks (linear rows of Bt)
      int c = tid + t * 128;
      int row = c >> 2, cc = c & 3;
      async_cp16(sB + c * 8, Bt + (size_t)(c0 + row) * Kc + k0 + cc * 8);
    }
    async_wait();
    __syncthreads();
    v16bf fa0 = ld_fragA(sA + (wr + 0  + lr) * 32, half, 0);
    v16bf fa1 = ld_fragA(sA + (wr + 16 + lr) * 32, half, 0);
    v16bf fb0 = ld_fragB(sB + (wc + 0  + lr) * 32, half, 0);
    v16bf fb1 = ld_fragB(sB + (wc + 16 + lr) * 32, half, 0);
    a00 = wmma_bf16(fa0, fb0, a00);
    a01 = wmma_bf16(fa0, fb1, a01);
    a10 = wmma_bf16(fa1, fb0, a10);
    a11 = wmma_bf16(fa1, fb1, a11);
  }
#pragma unroll
  for (int p = 0; p < 8; ++p) {
    int ri = p + 8 * half;
    gemm_epi(mode, r0 + wr + 0  + ri, c0 + wc + 0  + lr, a00[p], rwb, rrb, qa, qb, ka, va, ra, aout);
    gemm_epi(mode, r0 + wr + 0  + ri, c0 + wc + 16 + lr, a01[p], rwb, rrb, qa, qb, ka, va, ra, aout);
    gemm_epi(mode, r0 + wr + 16 + ri, c0 + wc + 0  + lr, a10[p], rwb, rrb, qa, qb, ka, va, ra, aout);
    gemm_epi(mode, r0 + wr + 16 + ri, c0 + wc + 16 + lr, a11[p], rwb, rrb, qa, qb, ka, va, ra, aout);
  }
}

// ---------------- fused flash attention with analytic rel-shift ----------------
// grid (T/64, N, B), 128 threads; wave w owns rows i0+16w .. +15.
__global__ __launch_bounds__(128) void attn_kernel(
    const __bf16* __restrict__ qa, const __bf16* __restrict__ qb,
    const __bf16* __restrict__ kk_, const __bf16* __restrict__ vv_,
    const __bf16* __restrict__ rr_, __bf16* __restrict__ av) {
  __shared__ __bf16 sQa[64 * 64], sQb[64 * 64];  // [i][d]
  __shared__ __bf16 sK[32 * 64];                 // [jj][d]
  __shared__ __bf16 sV[64 * 32];                 // [d][jj] (global already transposed)
  __shared__ __bf16 sR[96 * 64];                 // [mm][d]
  __shared__ float  sS[4][16 * 32];
  __shared__ float  sBD[4][16 * 48];
  __shared__ __bf16 sP[4][16 * 32];
  __shared__ float  sScale[4][16];

  int tid = threadIdx.x, wave = tid >> 5, lane = tid & 31, lr = lane & 15, half = lane >> 4;
  int i0 = blockIdx.x * 64;
  int n = blockIdx.y, b = blockIdx.z;
  const __bf16* qa_g = qa + ((size_t)(b * 16 + n) * 1024 + i0) * 64;
  const __bf16* qb_g = qb + ((size_t)(b * 16 + n) * 1024 + i0) * 64;
  const __bf16* k_g = kk_ + (size_t)(b * 16 + n) * 2048 * 64;
  const __bf16* v_g = vv_ + (size_t)(b * 16 + n) * 64 * 2048;   // [d][j]
  const __bf16* r_g = rr_ + (size_t)n * 2048 * 64;

  {                                              // stage Q tiles once (async DMA)
#pragma unroll
    for (int t = 0; t < 4; ++t) {
      int c = tid + t * 128;                     // 512 chunks each
      async_cp16(sQa + c * 8, qa_g + c * 8);
      async_cp16(sQb + c * 8, qb_g + c * 8);
    }
    async_wait();
  }
  __syncthreads();

  v8f o0 = {0,0,0,0,0,0,0,0}, o1 = o0, o2 = o0, o3 = o0;
  float m_run = -1e30f, l_run = 0.f;
  int iw0 = i0 + wave * 16;
  int i_g = iw0 + lr;
  int jmax = i0 + 64 + TMEM;                     // exclusive, multiple of 32

  for (int j0 = 0; j0 < jmax; j0 += 32) {
    __syncthreads();
#if HAVE_TDM
    if (wave == 0) {                             // TDM: one descriptor per tile
      int mbase = j0 + 960 - i0;                 // >= 0; rows beyond tensor -> zero (masked)
      tdm_load_2d(k_g + (size_t)j0 * 64, lds_addr_of(sK), 512, 1, 512, 1, 512);
      tdm_load_2d(v_g + j0, lds_addr_of(sV), 8, 64, 8, 64, 512);
      tdm_load_2d(r_g + (size_t)mbase * 64, lds_addr_of(sR),
                  16, 96, 16, (u32)(2048 - mbase), 16);
      tdm_wait();
    }
#else
    {                                            // K tile: 256 chunks, contiguous
#pragma unroll
      for (int t = 0; t < 2; ++t) {
        int c = tid + t * 128;
        async_cp16(sK + c * 8, k_g + (size_t)j0 * 64 + c * 8);
      }
    }
    {                                            // V tile: rows d, 64B per row
#pragma unroll
      for (int t = 0; t < 2; ++t) {
        int c = tid + t * 128;
        int d = c >> 2, cc = c & 3;
        async_cp16(sV + c * 8, v_g + (size_t)d * 2048 + j0 + cc * 8);
      }
    }
    {                                            // R rows [mbase .. mbase+95]
      int mbase = j0 + 960 - i0;
#pragma unroll
      for (int t = 0; t < 6; ++t) {
        int c = tid + t * 128;                   // 768 chunks
        int mm = c >> 3, cc = c & 7;
        int m = mbase + mm; m = m < 0 ? 0 : (m > 2047 ? 2047 : m);
        async_cp16(sR + c * 8, r_g + (size_t)m * 64 + cc * 8);
      }
    }
    async_wait();
#endif
    __syncthreads();

    // AC = Q_ac (16x64) x K^T (64x32)
    v8f acS0 = {0,0,0,0,0,0,0,0}, acS1 = acS0;
    {
      v16bf q0 = ld_fragA(sQa + (wave * 16 + lr) * 64, half, 0);
      v16bf q1 = ld_fragA(sQa + (wave * 16 + lr) * 64, half, 32);
      v16bf b00 = ld_fragB(sK + (0  + lr) * 64, half, 0);
      v16bf b01 = ld_fragB(sK + (0  + lr) * 64, half, 32);
      v16bf b10 = ld_fragB(sK + (16 + lr) * 64, half, 0);
      v16bf b11 = ld_fragB(sK + (16 + lr) * 64, half, 32);
      acS0 = wmma_bf16(q0, b00, acS0); acS0 = wmma_bf16(q1, b01, acS0);
      acS1 = wmma_bf16(q0, b10, acS1); acS1 = wmma_bf16(q1, b11, acS1);
    }
    // BD block 16x48: c -> R row (48 - 16w + c) in sR
    v8f bd0 = {0,0,0,0,0,0,0,0}, bd1 = bd0, bd2 = bd0;
    {
      v16bf q0 = ld_fragA(sQb + (wave * 16 + lr) * 64, half, 0);
      v16bf q1 = ld_fragA(sQb + (wave * 16 + lr) * 64, half, 32);
      int rb = 48 - wave * 16;
      v16bf r00 = ld_fragB(sR + (rb + 0  + lr) * 64, half, 0);
      v16bf r01 = ld_fragB(sR + (rb + 0  + lr) * 64, half, 32);
      v16bf r10 = ld_fragB(sR + (rb + 16 + lr) * 64, half, 0);
      v16bf r11 = ld_fragB(sR + (rb + 16 + lr) * 64, half, 32);
      v16bf r20 = ld_fragB(sR + (rb + 32 + lr) * 64, half, 0);
      v16bf r21 = ld_fragB(sR + (rb + 32 + lr) * 64, half, 32);
      bd0 = wmma_bf16(q0, r00, bd0); bd0 = wmma_bf16(q1, r01, bd0);
      bd1 = wmma_bf16(q0, r10, bd1); bd1 = wmma_bf16(q1, r11, bd1);
      bd2 = wmma_bf16(q0, r20, bd2); bd2 = wmma_bf16(q1, r21, bd2);
    }
    // dump per-wave score tiles to LDS (C/D layout: lane->col, vgpr->row)
#pragma unroll
    for (int p = 0; p < 8; ++p) {
      int rrow = p + 8 * half;
      sS[wave][rrow * 32 + 0  + lr] = acS0[p];
      sS[wave][rrow * 32 + 16 + lr] = acS1[p];
      sBD[wave][rrow * 48 + 0  + lr] = bd0[p];
      sBD[wave][rrow * 48 + 16 + lr] = bd1[p];
      sBD[wave][rrow * 48 + 32 + lr] = bd2[p];
    }
    asm volatile("s_wait_dscnt 0" ::: "memory");

    // online softmax: lane handles row lr, cols half*16..+15
    float mloc = -1e30f;
    float sv[16];
#pragma unroll
    for (int q = 0; q < 16; ++q) {
      int jl = half * 16 + q;
      int j = j0 + jl;
      float s = (sS[wave][lr * 32 + jl] + sBD[wave][lr * 48 + jl + 15 - lr]) * 0.125f;
      if (j > i_g + TMEM) s = -1e30f;
      sv[q] = s;
      mloc = fmaxf(mloc, s);
    }
    float mnew = fmaxf(m_run, fmaxf(mloc, __shfl_xor(mloc, 16)));
    float fac = __expf(m_run - mnew);
    float lloc = 0.f;
#pragma unroll
    for (int q = 0; q < 16; ++q) {
      float pe = __expf(sv[q] - mnew);
      lloc += pe;
      sP[wave][lr * 32 + half * 16 + q] = (__bf16)pe;
    }
    l_run = l_run * fac + lloc + __shfl_xor(lloc, 16);
    m_run = mnew;
    if (half == 0) sScale[wave][lr] = fac;
    asm volatile("s_wait_dscnt 0" ::: "memory");

    // rescale O, then O += P(16x32) x V(32x64)
    v16bf ap = ld_fragA(sP[wave] + lr * 32, half, 0);
    v16bf bv0 = ld_fragB(sV + (0  + lr) * 32, half, 0);
    v16bf bv1 = ld_fragB(sV + (16 + lr) * 32, half, 0);
    v16bf bv2 = ld_fragB(sV + (32 + lr) * 32, half, 0);
    v16bf bv3 = ld_fragB(sV + (48 + lr) * 32, half, 0);
#pragma unroll
    for (int p = 0; p < 8; ++p) {
      float f = sScale[wave][p + 8 * half];
      o0[p] *= f; o1[p] *= f; o2[p] *= f; o3[p] *= f;
    }
    o0 = wmma_bf16(ap, bv0, o0);
    o1 = wmma_bf16(ap, bv1, o1);
    o2 = wmma_bf16(ap, bv2, o2);
    o3 = wmma_bf16(ap, bv3, o3);
  }

  if (half == 0) sScale[wave][lr] = 1.0f / l_run;
  asm volatile("s_wait_dscnt 0" ::: "memory");
#pragma unroll
  for (int p = 0; p < 8; ++p) {
    float f = sScale[wave][p + 8 * half];
    int row_i = iw0 + p + 8 * half;
    size_t base = ((size_t)row_i * 2 + blockIdx.z) * 1024 + blockIdx.y * 64;
    av[base + 0  + lr] = (__bf16)(o0[p] * f);
    av[base + 16 + lr] = (__bf16)(o1[p] * f);
    av[base + 32 + lr] = (__bf16)(o2[p] * f);
    av[base + 48 + lr] = (__bf16)(o3[p] * f);
  }
}

// ---------------- LayerNorm: one row (D=1024) per 256-thread block ----------------
__global__ __launch_bounds__(256) void ln_kernel(const float* __restrict__ x,
                                                 const float* __restrict__ gamma,
                                                 const float* __restrict__ beta,
                                                 float* __restrict__ out) {
  __shared__ float red[8];
  int r = blockIdx.x, tid = threadIdx.x;
  const float* xr = x + (size_t)r * 1024;
  float v4[4];
  float s = 0.f;
#pragma unroll
  for (int t = 0; t < 4; ++t) { v4[t] = xr[tid + t * 256]; s += v4[t]; }
#pragma unroll
  for (int off = 16; off > 0; off >>= 1) s += __shfl_xor(s, off);
  if ((tid & 31) == 0) red[tid >> 5] = s;
  __syncthreads();
  float tot = 0.f;
#pragma unroll
  for (int w = 0; w < 8; ++w) tot += red[w];
  float mean = tot * (1.0f / 1024.0f);
  float s2 = 0.f;
#pragma unroll
  for (int t = 0; t < 4; ++t) { float d = v4[t] - mean; s2 += d * d; }
#pragma unroll
  for (int off = 16; off > 0; off >>= 1) s2 += __shfl_xor(s2, off);
  __syncthreads();
  if ((tid & 31) == 0) red[tid >> 5] = s2;
  __syncthreads();
  float tv = 0.f;
#pragma unroll
  for (int w = 0; w < 8; ++w) tv += red[w];
  float rstd = rsqrtf(tv * (1.0f / 1024.0f) + 1e-5f);
#pragma unroll
  for (int t = 0; t < 4; ++t) {
    int d = tid + t * 256;
    out[(size_t)r * 1024 + d] = (v4[t] - mean) * rstd * gamma[d] + beta[d];
  }
}

extern "C" void kernel_launch(void* const* d_in, const int* in_sizes, int n_in,
                              void* d_out, int out_size, void* d_ws, size_t ws_size,
                              hipStream_t stream) {
  (void)in_sizes; (void)n_in; (void)out_size; (void)ws_size;
  const float* inpts = (const float*)d_in[0];
  const float* pos   = (const float*)d_in[1];
  const float* mems  = (const float*)d_in[2];
  const float* Wqkv  = (const float*)d_in[4];
  const float* Wr    = (const float*)d_in[5];
  const float* Wout  = (const float*)d_in[6];
  const float* rwb   = (const float*)d_in[7];
  const float* rrb   = (const float*)d_in[8];
  const float* gamma = (const float*)d_in[9];
  const float* beta  = (const float*)d_in[10];
  float* out = (float*)d_out;

  char* ws = (char*)d_ws;
  size_t off = 0;
  auto alloc = [&](size_t bytes) {
    char* p = ws + off;
    off += (bytes + 255) & ~(size_t)255;
    return p;
  };
  __bf16* Wqkv_t = (__bf16*)alloc((size_t)1024 * 3072 * 2);   // [n][k]
  __bf16* Wr_t   = (__bf16*)alloc((size_t)1024 * 1024 * 2);   // [n][k]
  __bf16* Wout_t = (__bf16*)alloc((size_t)1024 * 1024 * 2);   // [n][k]
  __bf16* pos_h  = (__bf16*)alloc((size_t)2048 * 1024 * 2);
  __bf16* cat_h  = (__bf16*)alloc((size_t)4096 * 1024 * 2);
  __bf16* qa     = (__bf16*)alloc((size_t)2 * 16 * 1024 * 64 * 2);
  __bf16* qb     = (__bf16*)alloc((size_t)2 * 16 * 1024 * 64 * 2);
  __bf16* ka     = (__bf16*)alloc((size_t)2 * 16 * 2048 * 64 * 2);
  __bf16* va     = (__bf16*)alloc((size_t)2 * 16 * 64 * 2048 * 2);  // [b][n][d][j]
  __bf16* ra     = (__bf16*)alloc((size_t)16 * 2048 * 64 * 2);
  __bf16* av     = (__bf16*)alloc((size_t)2048 * 1024 * 2);
  float*  aout   = (float*)alloc((size_t)2048 * 1024 * 4);

  // converts (weights transposed once -> linear async staging in GEMMs)
  cvt_t_kernel<<<(1024 * 3072) / 256, 256, 0, stream>>>(Wqkv, Wqkv_t, 1024, 3072);
  cvt_t_kernel<<<(1024 * 1024) / 256, 256, 0, stream>>>(Wr, Wr_t, 1024, 1024);
  cvt_t_kernel<<<(1024 * 1024) / 256, 256, 0, stream>>>(Wout, Wout_t, 1024, 1024);
  cvt_kernel<<<(2048 * 1024) / 256, 256, 0, stream>>>(pos, pos_h, 2048 * 1024);
  cat_kernel<<<(4096 * 1024) / 256, 256, 0, stream>>>(mems, inpts, cat_h);

  // w_heads = cat @ W_qkv  (M=4096, N=3072) with q/k/v scatter + bias epilogue
  gemm_bf16_kernel<<<dim3(48, 64), 128, 0, stream>>>(
      cat_h, Wqkv_t, 0, rwb, rrb, qa, qb, ka, va, nullptr, nullptr);
  // r_head_k = pos @ W_r  (M=2048, N=1024)
  gemm_bf16_kernel<<<dim3(16, 32), 128, 0, stream>>>(
      pos_h, Wr_t, 1, nullptr, nullptr, nullptr, nullptr, nullptr, nullptr, ra, nullptr);

  // fused flash attention with analytic rel-shift (TDM-staged K/V/R tiles)
  attn_kernel<<<dim3(16, 16, 2), 128, 0, stream>>>(qa, qb, ka, va, ra, av);

  // attn_out = attn_vec @ W_out  (M=2048, N=1024)
  gemm_bf16_kernel<<<dim3(16, 32), 128, 0, stream>>>(
      av, Wout_t, 2, nullptr, nullptr, nullptr, nullptr, nullptr, nullptr, nullptr, aout);

  // LayerNorm -> output (T, B, D)
  ln_kernel<<<2048, 256, 0, stream>>>(aout, gamma, beta, out);
}
